// CausalSelfAttention_33054068310257
// MI455X (gfx1250) — compile-verified, matrix operand order checked by reading, not run on hardware
//
#include <hip/hip_runtime.h>
#include <hip/hip_bf16.h>
#include <math.h>

// ---------------------------------------------------------------------------
// CDNA5 (gfx1250) causal self-attention with sparsemax-K and XSA correction.
// Matrix math: v_wmma_f32_16x16x32_bf16 (fp32 accumulate).
// Flash K/V tiles: Tensor Data Mover (tensor_load_to_lds) double-buffered in
// LDS, synchronized with s_wait_tensorcnt (1 wave per WG so TENSORcnt works).
// ---------------------------------------------------------------------------

typedef __bf16 bf16;
typedef __attribute__((ext_vector_type(16))) __bf16 v16bf;
typedef __attribute__((ext_vector_type(8)))  __bf16 v8bf;
typedef __attribute__((ext_vector_type(8)))  float  v8f;
typedef __attribute__((ext_vector_type(4)))  unsigned v4u;
typedef __attribute__((ext_vector_type(4)))  int   v4i;
typedef __attribute__((ext_vector_type(8)))  int   v8i;

#define BB 4
#define TT 2048
#define CC 1024
#define HH 16
#define DD 64
#define MM (BB * TT)       // 8192
#define NEG_INF_F (-1e30f)

// ---- fragment helpers ------------------------------------------------------
// 16-bit A-matrix 16x32 layout (ISA 7.12.2): lanes 0-15 hold K=0..7 / K=16..23,
// lanes 16-31 hold K=8..15 / K=24..31 -> two contiguous 16B loads per lane.
__device__ inline const bf16* frag_ptr(const bf16* base, int ld, int row0, int k0, int lane) {
    return base + (long)(row0 + (lane & 15)) * ld + k0 + ((lane >> 4) << 3);
}

__device__ inline v16bf load_frag(const bf16* p) {
    v8bf lo = *(const v8bf*)(p);
    v8bf hi = *(const v8bf*)(p + 16);
    v16bf r;
#pragma unroll
    for (int i = 0; i < 8; ++i) { r[i] = lo[i]; r[i + 8] = hi[i]; }
    return r;
}

__device__ inline v8f v8f_zero() {
    v8f z = {0.f, 0.f, 0.f, 0.f, 0.f, 0.f, 0.f, 0.f};
    return z;
}

__device__ inline v8f wmma_bf16(v16bf a, v16bf b, v8f c) {
    return __builtin_amdgcn_wmma_f32_16x16x32_bf16(false, a, false, b, (short)0, c, false, false);
}

// reductions across a 16-lane half-wave (C/D layout: lanes 0-15 rows j,
// lanes 16-31 rows j+8; xor masks 1..8 stay within each half)
__device__ inline float half16_max(float v) {
#pragma unroll
    for (int m = 8; m >= 1; m >>= 1) v = fmaxf(v, __shfl_xor(v, m, 32));
    return v;
}
__device__ inline float half16_sum(float v) {
#pragma unroll
    for (int m = 8; m >= 1; m >>= 1) v += __shfl_xor(v, m, 32);
    return v;
}

// ---- Tensor Data Mover: 2D 64x64 (2-byte elems) tile -> LDS ---------------
// D# group0: count=1 | lds_addr | global_addr[56:0] | type=2
// D# group1: data_size=1 (2B), tensor_dim0/1 = 64, tile_dim0/1 = 64,
//            tensor_dim0_stride = row_stride_elems. Tile never exceeds the
//            tensor here so OOB fields only need to cover the tile.
__device__ inline void tdm_load_64x64_b16(unsigned lds_byte_addr, const bf16* gsrc,
                                          unsigned row_stride_elems) {
    unsigned long long ga = (unsigned long long)(const void*)gsrc;
    v4u g0;
    g0[0] = 1u;                                             // count=1
    g0[1] = lds_byte_addr;                                  // lds_addr
    g0[2] = (unsigned)(ga & 0xffffffffu);                   // global_addr lo
    g0[3] = (unsigned)((ga >> 32) & 0x01ffffffu) | (2u << 30); // addr hi | type=2
    v8i g1;
    g1[0] = (int)(1u << 16);                // data_size=1 (2 bytes/elem)
    g1[1] = (int)(64u << 16);               // tensor_dim0[15:0] = 64
    g1[2] = (int)(64u << 16);               // tensor_dim1[15:0] = 64
    g1[3] = (int)(64u << 16);               // tile_dim0 = 64
    g1[4] = 64;                             // tile_dim1 = 64, tile_dim2 = 0
    g1[5] = (int)row_stride_elems;          // tensor_dim0_stride (low 32)
    g1[6] = 0;
    g1[7] = 0;
    v4i z4 = {0, 0, 0, 0};
#if defined(__clang_major__) && (__clang_major__ >= 23)
    v8i z8 = {0, 0, 0, 0, 0, 0, 0, 0};
    __builtin_amdgcn_tensor_load_to_lds(g0, g1, z4, z4, z8, 0);
#else
    __builtin_amdgcn_tensor_load_to_lds(g0, g1, z4, z4, 0);
#endif
}

// ---- stage 1: fp32 -> bf16 conversions ------------------------------------
__global__ void cvt_f32_bf16(const float* __restrict__ src, bf16* __restrict__ dst, long n) {
    long i = (long)blockIdx.x * blockDim.x + threadIdx.x;
    if (i < n) dst[i] = (bf16)src[i];
}

__global__ void cvt_transpose_bf16(const float* __restrict__ W, bf16* __restrict__ Wt, int K, int N) {
    long i = (long)blockIdx.x * blockDim.x + threadIdx.x;
    if (i < (long)K * N) {
        int kk = (int)(i / N), n = (int)(i % N);
        Wt[(long)n * K + kk] = (bf16)W[i];
    }
}

// ---- wave-level 64x64 GEMM core (A [M,K] bf16, Bt [N,K] bf16) -------------
__device__ inline void wave_gemm_64x64(const bf16* __restrict__ A, const bf16* __restrict__ Bt,
                                       int K, int mbase, int nbase, int lane, v8f c[4][4]) {
#pragma unroll
    for (int mi = 0; mi < 4; ++mi)
#pragma unroll
        for (int ni = 0; ni < 4; ++ni) c[mi][ni] = v8f_zero();

    for (int k0 = 0; k0 < K; k0 += 32) {
        v16bf a[4], b[4];
#pragma unroll
        for (int mi = 0; mi < 4; ++mi) a[mi] = load_frag(frag_ptr(A, K, mbase + mi * 16, k0, lane));
#pragma unroll
        for (int ni = 0; ni < 4; ++ni) b[ni] = load_frag(frag_ptr(Bt, K, nbase + ni * 16, k0, lane));
#pragma unroll
        for (int mi = 0; mi < 4; ++mi)
#pragma unroll
            for (int ni = 0; ni < 4; ++ni)
                c[mi][ni] = wmma_bf16(a[mi], b[ni], c[mi][ni]);
    }
}

// ---- stage 2: QKV GEMM + bias + scatter into head layouts -----------------
__global__ __launch_bounds__(128) void qkv_gemm_kernel(
    const bf16* __restrict__ xb, const bf16* __restrict__ Wat, const float* __restrict__ bias,
    bf16* __restrict__ qb, float* __restrict__ kf, float* __restrict__ vf, bf16* __restrict__ vt) {
    int lane = threadIdx.x & 31;
    int wave = threadIdx.x >> 5;
    int Mb = blockIdx.y * 128 + (wave >> 1) * 64;
    int Nb = blockIdx.x * 128 + (wave & 1) * 64;

    v8f c[4][4];
    wave_gemm_64x64(xb, Wat, CC, Mb, Nb, lane, c);

#pragma unroll
    for (int mi = 0; mi < 4; ++mi)
#pragma unroll
        for (int ni = 0; ni < 4; ++ni) {
            int col = Nb + ni * 16 + (lane & 15);
            float bv = bias[col];
#pragma unroll
            for (int j = 0; j < 8; ++j) {
                int row = Mb + mi * 16 + ((lane >> 4) << 3) + j;
                float val = c[mi][ni][j] + bv;
                int bidx = row >> 11, t = row & (TT - 1);
                if (col < CC) {
                    int h = col >> 6, d = col & 63;
                    qb[((long)(bidx * HH + h) * TT + t) * DD + d] = (bf16)(val * 0.125f);
                } else if (col < 2 * CC) {
                    int n = col - CC, h = n >> 6, d = n & 63;
                    kf[((long)(bidx * HH + h) * TT + t) * DD + d] = val;
                } else {
                    int n = col - 2 * CC, h = n >> 6, d = n & 63;
                    vf[((long)(bidx * HH + h) * TT + t) * DD + d] = val;
                    vt[((long)(bidx * HH + h) * DD + d) * TT + t] = (bf16)val;
                }
            }
        }
}

// ---- stage 3: sparsemax over D=64 (one wave per row, bisection on tau) ----
__global__ __launch_bounds__(256) void sparsemax_kernel(
    const float* __restrict__ kf, bf16* __restrict__ kb, int rows) {
    int wid = (int)((blockIdx.x * (long)blockDim.x + threadIdx.x) >> 5);
    int lane = threadIdx.x & 31;
    if (wid >= rows) return;
    const float* z = kf + (long)wid * DD;
    float z0 = z[lane], z1 = z[lane + 32];
    float m = fmaxf(z0, z1);
#pragma unroll
    for (int o = 16; o >= 1; o >>= 1) m = fmaxf(m, __shfl_xor(m, o, 32));
    float lo = m - 1.0f, hi = m;
    for (int it = 0; it < 30; ++it) {
        float tau = 0.5f * (lo + hi);
        float s = fmaxf(z0 - tau, 0.f) + fmaxf(z1 - tau, 0.f);
#pragma unroll
        for (int o = 16; o >= 1; o >>= 1) s += __shfl_xor(s, o, 32);
        if (s > 1.0f) lo = tau; else hi = tau;
    }
    float tau = 0.5f * (lo + hi);
    bf16* o = kb + (long)wid * DD;
    o[lane]      = (bf16)fmaxf(z0 - tau, 0.f);
    o[lane + 32] = (bf16)fmaxf(z1 - tau, 0.f);
}

// ---- stage 4: flash attention (causal) + XSA correction -------------------
// One wave per 64 query rows. 64-key chunks; K and V^T tiles DMA'd to LDS by
// the Tensor Data Mover (double-buffered, s_wait_tensorcnt); online softmax;
// P transposed through LDS into the A-fragment layout for PV.
__global__ __launch_bounds__(32) void flash_xsa_kernel(
    const bf16* __restrict__ qb, const bf16* __restrict__ kb,
    const bf16* __restrict__ vt, const float* __restrict__ vf,
    bf16* __restrict__ yb) {
    __shared__ __attribute__((aligned(16))) bf16 Kl[2][64 * 64];  // [key][d]
    __shared__ __attribute__((aligned(16))) bf16 Vl[2][64 * 64];  // [d][key]
    __shared__ __attribute__((aligned(16))) bf16 P[64 * 64];      // [row][key]
    int lane = threadIdx.x;
    int m0 = blockIdx.x * 64;
    int bh = blockIdx.y;                 // b*H + h
    int b = bh >> 4, h = bh & 15;

    const bf16* qbase = qb + (long)bh * TT * DD;
    const bf16* kbase = kb + (long)bh * TT * DD;
    const bf16* vtb   = vt + (long)bh * DD * TT;
    const float* vfb  = vf + (long)bh * TT * DD;

    // Q fragments: 4 row-tiles x 2 K-steps (q pre-scaled by 1/sqrt(D))
    v16bf qa[4][2];
#pragma unroll
    for (int mi = 0; mi < 4; ++mi) {
        qa[mi][0] = load_frag(frag_ptr(qbase, DD, m0 + mi * 16, 0, lane));
        qa[mi][1] = load_frag(frag_ptr(qbase, DD, m0 + mi * 16, 32, lane));
    }

    float mrow[4][8], lrow[4][8];
    v8f acc[4][4];
#pragma unroll
    for (int mi = 0; mi < 4; ++mi) {
#pragma unroll
        for (int j = 0; j < 8; ++j) { mrow[mi][j] = NEG_INF_F; lrow[mi][j] = 0.f; }
#pragma unroll
        for (int dt = 0; dt < 4; ++dt) acc[mi][dt] = v8f_zero();
    }

    int col_lo = lane & 15;
    int row_hi = (lane >> 4) << 3;
    int nchunks = blockIdx.x + 1;        // causal: keys < m0 + 64

    // prime the TDM pipeline: chunk 0 into buffer 0
    unsigned k0_lds = (unsigned)(unsigned long long)(const void*)&Kl[0][0];
    unsigned v0_lds = (unsigned)(unsigned long long)(const void*)&Vl[0][0];
    unsigned k1_lds = (unsigned)(unsigned long long)(const void*)&Kl[1][0];
    unsigned v1_lds = (unsigned)(unsigned long long)(const void*)&Vl[1][0];
    tdm_load_64x64_b16(k0_lds, kbase, DD);          // K rows: contiguous 8KB
    tdm_load_64x64_b16(v0_lds, vtb, TT);            // V^T rows: stride T

    for (int c = 0; c < nchunks; ++c) {
        int n0 = c * 64;
        int cur = c & 1;
        if (c + 1 < nchunks) {
            int n1 = n0 + 64;
            tdm_load_64x64_b16(cur ? k0_lds : k1_lds, kbase + (long)n1 * DD, DD);
            tdm_load_64x64_b16(cur ? v0_lds : v1_lds, vtb + n1, TT);
            __builtin_amdgcn_s_wait_tensorcnt(2);   // current pair landed
        } else {
            __builtin_amdgcn_s_wait_tensorcnt(0);
        }
        const bf16* Kc = &Kl[cur][0];
        const bf16* Vc = &Vl[cur][0];

        // ---- S = Q K^T per row-tile, online softmax, P -> LDS ----
#pragma unroll
        for (int mi = 0; mi < 4; ++mi) {
            v8f s[4];
#pragma unroll
            for (int t = 0; t < 4; ++t) {
                v16bf kb0 = load_frag(frag_ptr(Kc, DD, 16 * t, 0, lane));
                v16bf kb1 = load_frag(frag_ptr(Kc, DD, 16 * t, 32, lane));
                v8f z = v8f_zero();
                z = wmma_bf16(qa[mi][0], kb0, z);
                z = wmma_bf16(qa[mi][1], kb1, z);
                s[t] = z;
            }
#pragma unroll
            for (int j = 0; j < 8; ++j) {
                int row = m0 + mi * 16 + row_hi + j;
                float mx = NEG_INF_F;
#pragma unroll
                for (int t = 0; t < 4; ++t) {
                    if (n0 + 16 * t + col_lo > row) s[t][j] = NEG_INF_F;
                    mx = fmaxf(mx, s[t][j]);
                }
                mx = half16_max(mx);
                float nm = fmaxf(mrow[mi][j], mx);
                float scale = __expf(mrow[mi][j] - nm);
                mrow[mi][j] = nm;
                float sumj = 0.f;
#pragma unroll
                for (int t = 0; t < 4; ++t) {
                    float p = __expf(s[t][j] - nm);
                    sumj += p;
                    P[(mi * 16 + row_hi + j) * 64 + 16 * t + col_lo] = (bf16)p;
                }
                sumj = half16_sum(sumj);
                lrow[mi][j] = lrow[mi][j] * scale + sumj;
#pragma unroll
                for (int dt = 0; dt < 4; ++dt) acc[mi][dt][j] *= scale;
            }
        }
        __syncthreads();                 // 1-wave WG: S_NOP barrier, orders LDS

        // ---- PV: A = P (LDS-transposed fragments), B = V^T rows ----
        v16bf pa[4][2];
#pragma unroll
        for (int mi = 0; mi < 4; ++mi) {
            pa[mi][0] = load_frag(frag_ptr(P, 64, mi * 16, 0, lane));
            pa[mi][1] = load_frag(frag_ptr(P, 64, mi * 16, 32, lane));
        }
#pragma unroll
        for (int dt = 0; dt < 4; ++dt) {
            v16bf vb0 = load_frag(frag_ptr(Vc, 64, dt * 16, 0, lane));
            v16bf vb1 = load_frag(frag_ptr(Vc, 64, dt * 16, 32, lane));
#pragma unroll
            for (int mi = 0; mi < 4; ++mi) {
                acc[mi][dt] = wmma_bf16(pa[mi][0], vb0, acc[mi][dt]);
                acc[mi][dt] = wmma_bf16(pa[mi][1], vb1, acc[mi][dt]);
            }
        }
        __syncthreads();
    }

    // ---- finalize: y = acc/l, then y -= (y.v_hat) v_hat ----
#pragma unroll
    for (int mi = 0; mi < 4; ++mi)
#pragma unroll
        for (int j = 0; j < 8; ++j) {
            int row = m0 + mi * 16 + row_hi + j;
            float inv = 1.0f / lrow[mi][j];
            float yv[4], vv[4], sv = 0.f, syv = 0.f;
#pragma unroll
            for (int dt = 0; dt < 4; ++dt) {
                float y = acc[mi][dt][j] * inv;
                float v = vfb[(long)row * DD + dt * 16 + col_lo];
                yv[dt] = y; vv[dt] = v;
                sv += v * v; syv += y * v;
            }
            sv = half16_sum(sv);
            syv = half16_sum(syv);
            float vn = fmaxf(sqrtf(sv), 1e-12f);
            float coef = syv / (vn * vn);
#pragma unroll
            for (int dt = 0; dt < 4; ++dt) {
                float yo = yv[dt] - coef * vv[dt];
                yb[((long)(b * TT + row)) * CC + h * DD + dt * 16 + col_lo] = (bf16)yo;
            }
        }
}

// ---- stage 5: output projection -------------------------------------------
__global__ __launch_bounds__(128) void proj_gemm_kernel(
    const bf16* __restrict__ yb, const bf16* __restrict__ Wpt,
    const float* __restrict__ bias, float* __restrict__ out) {
    int lane = threadIdx.x & 31;
    int wave = threadIdx.x >> 5;
    int Mb = blockIdx.y * 128 + (wave >> 1) * 64;
    int Nb = blockIdx.x * 128 + (wave & 1) * 64;

    v8f c[4][4];
    wave_gemm_64x64(yb, Wpt, CC, Mb, Nb, lane, c);

#pragma unroll
    for (int mi = 0; mi < 4; ++mi)
#pragma unroll
        for (int ni = 0; ni < 4; ++ni) {
            int col = Nb + ni * 16 + (lane & 15);
            float bv = bias[col];
#pragma unroll
            for (int j = 0; j < 8; ++j) {
                int row = Mb + mi * 16 + ((lane >> 4) << 3) + j;
                out[(long)row * CC + col] = c[mi][ni][j] + bv;
            }
        }
}

// ---------------------------------------------------------------------------
extern "C" void kernel_launch(void* const* d_in, const int* in_sizes, int n_in,
                              void* d_out, int out_size, void* d_ws, size_t ws_size,
                              hipStream_t stream) {
    (void)in_sizes; (void)n_in; (void)out_size; (void)ws_size;
    const float* x      = (const float*)d_in[0];
    const float* W_attn = (const float*)d_in[1];
    const float* b_attn = (const float*)d_in[2];
    const float* W_proj = (const float*)d_in[3];
    const float* b_proj = (const float*)d_in[4];
    float* out = (float*)d_out;

    char* ws = (char*)d_ws;
    size_t off = 0;
    auto carve = [&](size_t bytes) -> void* {
        void* p = ws + off;
        off += (bytes + 255) & ~(size_t)255;
        return p;
    };
    const long nX = (long)MM * CC;
    bf16*  xb  = (bf16*)carve(nX * 2);
    bf16*  Wat = (bf16*)carve((long)3 * CC * CC * 2);
    bf16*  Wpt = (bf16*)carve((long)CC * CC * 2);
    bf16*  qb  = (bf16*)carve(nX * 2);
    float* kf  = (float*)carve(nX * 4);
    bf16*  kbf = (bf16*)carve(nX * 2);
    float* vf  = (float*)carve(nX * 4);
    bf16*  vtb = (bf16*)carve(nX * 2);
    bf16*  yb  = (bf16*)carve(nX * 2);

    cvt_f32_bf16<<<(unsigned)((nX + 255) / 256), 256, 0, stream>>>(x, xb, nX);
    cvt_transpose_bf16<<<(unsigned)(((long)CC * 3 * CC + 255) / 256), 256, 0, stream>>>(
        W_attn, Wat, CC, 3 * CC);
    cvt_transpose_bf16<<<(unsigned)(((long)CC * CC + 255) / 256), 256, 0, stream>>>(
        W_proj, Wpt, CC, CC);

    qkv_gemm_kernel<<<dim3(3 * CC / 128, MM / 128), 128, 0, stream>>>(
        xb, Wat, b_attn, qb, kf, vf, vtb);

    sparsemax_kernel<<<(BB * HH * TT) / 8, 256, 0, stream>>>(kf, kbf, BB * HH * TT);

    flash_xsa_kernel<<<dim3(TT / 64, BB * HH), 32, 0, stream>>>(qb, kbf, vtb, vf, yb);

    proj_gemm_kernel<<<dim3(CC / 128, MM / 128), 128, 0, stream>>>(yb, Wpt, b_proj, out);
}
